// DecoderLayerWrapper_25890062861083
// MI455X (gfx1250) — matrix operand order repaired and checked by reading, not compiled, not run on hardware
//
#include <hip/hip_runtime.h>
#include <math.h>

// ---------------------------------------------------------------------------
// MI455X (gfx1250) single-token decoder layer.
// Bandwidth-bound: ~600 MB of HBM traffic (KV cache read + KV cache copy +
// weights) => ~26us floor at 23.3 TB/s. Compute (~60 MFLOP) is negligible, so
// GEMVs are fed through v_wmma_f32_16x16x32_f16 (f32 weights converted to f16
// in-register; B = hidden vector replicated across all 16 WMMA columns).
// Weight/KV streams are prefetched (global_prefetch_b8) to keep GL2 ahead of
// the demand loads.
// ---------------------------------------------------------------------------

typedef __attribute__((ext_vector_type(16))) _Float16 v16h;
typedef __attribute__((ext_vector_type(8)))  float    v8f;

#define NH      16
#define NKV     8
#define HD      128
#define DMODEL  1024
#define FFDIM   3072
#define EPS     1e-6f
#define CHUNK   1024          // attention positions per block
#define SCALE   0.08838834764831845f   // 128^-0.5

__device__ __forceinline__ float waveRedSum(float v) {
#pragma unroll
  for (int o = 16; o > 0; o >>= 1) v += __shfl_xor(v, o, 32);
  return v;
}

// ---------------------------------------------------------------------------
// RMSNorm over n elements (one block). Emits f32 and f16 copies.
// ---------------------------------------------------------------------------
__global__ void rmsnorm_kernel(const float* __restrict__ x,
                               const float* __restrict__ w,
                               float* __restrict__ out,
                               _Float16* __restrict__ outh, int n) {
  __shared__ float red[8];
  __shared__ float srms;
  float ssq = 0.f;
  for (int i = threadIdx.x; i < n; i += blockDim.x) { float v = x[i]; ssq += v * v; }
  ssq = waveRedSum(ssq);
  if ((threadIdx.x & 31) == 0) red[threadIdx.x >> 5] = ssq;
  __syncthreads();
  if (threadIdx.x == 0) {
    float t = 0.f;
    int nw = (blockDim.x + 31) >> 5;
    for (int i = 0; i < nw; ++i) t += red[i];
    srms = rsqrtf(t / (float)n + EPS);
  }
  __syncthreads();
  float rms = srms;
  for (int i = threadIdx.x; i < n; i += blockDim.x) {
    float v = x[i] * rms * w[i];
    out[i] = v;
    if (outh) outh[i] = (_Float16)v;
  }
}

// ---------------------------------------------------------------------------
// GEMV y[r] = sum_k W[r][k] * x[k]  (+ residual), via v_wmma_f32_16x16x32_f16.
// One wave computes a 16-row tile; K-loop unrolled x2 (two WMMAs / trip).
// 16-bit layouts used here (contiguous k order, same in A and B):
//   A lane l: row m = l%16, a[i] covers K = i + (l>=16 ? 16 : 0)
//   B lane l: b[i]  covers K = i + (l>=16 ? 16 : 0), independent of column N
//   D VGPR j: lanes 0-15 -> M=j, lanes 16-31 -> M=8+j
// R multiple of 16, K multiple of 64.
// ---------------------------------------------------------------------------
__device__ __forceinline__ v16h cvt_a_chunk(const float* p, int ka) {
  float4 f0 = *(const float4*)(p + ka);
  float4 f1 = *(const float4*)(p + ka + 4);
  float4 f2 = *(const float4*)(p + ka + 8);
  float4 f3 = *(const float4*)(p + ka + 12);
  v16h a;
  a[0]  = (_Float16)f0.x; a[1]  = (_Float16)f0.y;
  a[2]  = (_Float16)f0.z; a[3]  = (_Float16)f0.w;
  a[4]  = (_Float16)f1.x; a[5]  = (_Float16)f1.y;
  a[6]  = (_Float16)f1.z; a[7]  = (_Float16)f1.w;
  a[8]  = (_Float16)f2.x; a[9]  = (_Float16)f2.y;
  a[10] = (_Float16)f2.z; a[11] = (_Float16)f2.w;
  a[12] = (_Float16)f3.x; a[13] = (_Float16)f3.y;
  a[14] = (_Float16)f3.z; a[15] = (_Float16)f3.w;
  return a;
}

__global__ __launch_bounds__(256)
void gemv_wmma_kernel(const float* __restrict__ W,
                      const _Float16* __restrict__ x,
                      const float* __restrict__ residual,
                      float* __restrict__ y, int R, int K) {
  const int lane   = threadIdx.x & 31;
  const int wave   = threadIdx.x >> 5;
  const int tile   = blockIdx.x * (blockDim.x >> 5) + wave;
  const int rowBase = tile * 16;
  if (rowBase >= R) return;                  // wave-uniform (EXEC stays all-1s)
  const int hiHalf = (lane >= 16) ? 1 : 0;
  const int ka = hiHalf * 16;                // A K-offset for this half-wave
  const int kb = hiHalf * 16;                // B K-offset for this half-wave
  const int m = rowBase + (lane & 15);
  const float* Wrow = W + (size_t)m * (size_t)K;

  v8f acc = {};
  for (int c = 0; c < K; c += 64) {
    // Prefetch the next 64-float chunk of this row into GL2 (global_prefetch_b8).
    if (c + 64 < K) __builtin_prefetch(Wrow + c + 64, 0, 1);
    v16h a0 = cvt_a_chunk(Wrow + c, ka);
    v16h b0 = *(const v16h*)(x + c + kb);
    acc = __builtin_amdgcn_wmma_f32_16x16x32_f16(
        false, a0, false, b0, (short)0, acc, false, false);
    v16h a1 = cvt_a_chunk(Wrow + c + 32, ka);
    v16h b1 = *(const v16h*)(x + c + 32 + kb);
    acc = __builtin_amdgcn_wmma_f32_16x16x32_f16(
        false, a1, false, b1, (short)0, acc, false, false);
  }
  if ((lane & 15) == 0) {
    int mbase = rowBase + hiHalf * 8;
#pragma unroll
    for (int j = 0; j < 8; ++j) {
      float r = residual ? residual[mbase + j] : 0.f;
      y[mbase + j] = acc[j] + r;
    }
  }
}

// ---------------------------------------------------------------------------
// Per-head q/k RMSNorm + RoPE; write new k/v into cache tails.
// Grid: 16 q-heads | 8 k-heads | 8 v-heads. Block = 128 threads (one per d).
// ---------------------------------------------------------------------------
__global__ void qk_rope_kernel(const float* __restrict__ qp,
                               const float* __restrict__ kp,
                               const float* __restrict__ vp,
                               const float* __restrict__ qnw,
                               const float* __restrict__ knw,
                               const float* __restrict__ cosp,
                               const float* __restrict__ sinp,
                               float* __restrict__ qr,
                               float* __restrict__ k_new,
                               float* __restrict__ k_out,
                               float* __restrict__ v_out, int S) {
  const int b = blockIdx.x;
  const int d = threadIdx.x;   // 0..127
  __shared__ float sm[HD];
  __shared__ float red[4];
  __shared__ float srms;

  if (b < NH + NKV) {
    const float* vec = (b < NH) ? (qp + b * HD) : (kp + (b - NH) * HD);
    const float* wn  = (b < NH) ? qnw : knw;
    float val = vec[d];
    float ssq = waveRedSum(val * val);
    if ((d & 31) == 0) red[d >> 5] = ssq;
    __syncthreads();
    if (d == 0) srms = rsqrtf((red[0] + red[1] + red[2] + red[3]) / (float)HD + EPS);
    __syncthreads();
    float nv = val * srms * wn[d];
    sm[d] = nv;
    __syncthreads();
    float rot = (d < HD / 2) ? -sm[d + HD / 2] : sm[d - HD / 2];
    float out = nv * cosp[d] + rot * sinp[d];
    if (b < NH) {
      qr[b * HD + d] = out;
    } else {
      int h = b - NH;
      k_new[h * HD + d] = out;
      k_out[(size_t)h * (size_t)(S + 1) * HD + (size_t)S * HD + d] = out;
    }
  } else {
    int h = b - NH - NKV;
    v_out[(size_t)h * (size_t)(S + 1) * HD + (size_t)S * HD + d] = vp[h * HD + d];
  }
}

// ---------------------------------------------------------------------------
// Bulk KV cache copy: (NKV, S, HD) -> (NKV, S+1, HD) rows [0,S). float4.
// ---------------------------------------------------------------------------
__global__ void cache_copy_kernel(const float4* __restrict__ src,
                                  float4* __restrict__ dst, int S) {
  long idx = (long)blockIdx.x * blockDim.x + threadIdx.x;
  const long perHead = (long)S * (HD / 4);
  if (idx >= (long)NKV * perHead) return;
  long h = idx / perHead;
  long r = idx - h * perHead;
  dst[h * ((long)(S + 1) * (HD / 4)) + r] = src[idx];
}

// ---------------------------------------------------------------------------
// Flash-decoding partial attention. Grid (NKV, nchunk), block 256 (8 waves).
// Each wave: 128 positions, online softmax for the 2 grouped q-heads, then
// LDS combine across the 8 waves -> one partial {acc[128], m, l} per
// (head, g, chunk) in scratch (stride 132 floats).
// ---------------------------------------------------------------------------
__global__ __launch_bounds__(256)
void attn_partial_kernel(const float* __restrict__ kc,
                         const float* __restrict__ vc,
                         const float* __restrict__ qr,
                         const float* __restrict__ mask,
                         float* __restrict__ part, int S, int nchunk) {
  const int h     = blockIdx.x;
  const int chunk = blockIdx.y;
  const int lane  = threadIdx.x & 31;
  const int wave  = threadIdx.x >> 5;

  float4 q0 = *(const float4*)(qr + (h * 2 + 0) * HD + lane * 4);
  float4 q1 = *(const float4*)(qr + (h * 2 + 1) * HD + lane * 4);
  const float* kh = kc + (size_t)h * (size_t)S * HD;
  const float* vh = vc + (size_t)h * (size_t)S * HD;

  float m0 = -1e30f, l0 = 0.f, m1 = -1e30f, l1 = 0.f;
  float4 a0 = {0, 0, 0, 0}, a1 = {0, 0, 0, 0};

  const int perWave = CHUNK / 8;
  int s0 = chunk * CHUNK + wave * perWave;
  for (int s = s0; s < s0 + perWave && s < S; ++s) {
    // Prefetch K/V rows 4 positions ahead (stays within the block's chunk
    // most of the time; harmless GL2 warm-up otherwise).
    if (s + 4 < S) {
      __builtin_prefetch(kh + (size_t)(s + 4) * HD + lane * 4, 0, 1);
      __builtin_prefetch(vh + (size_t)(s + 4) * HD + lane * 4, 0, 1);
    }
    float4 kv = *(const float4*)(kh + (size_t)s * HD + lane * 4);
    float d0 = kv.x * q0.x + kv.y * q0.y + kv.z * q0.z + kv.w * q0.w;
    float d1 = kv.x * q1.x + kv.y * q1.y + kv.z * q1.z + kv.w * q1.w;
    d0 = waveRedSum(d0);
    d1 = waveRedSum(d1);
    float msk = mask[s];
    float sc0 = d0 * SCALE + msk;
    float sc1 = d1 * SCALE + msk;
    float4 vv = *(const float4*)(vh + (size_t)s * HD + lane * 4);
    {
      float nm = fmaxf(m0, sc0);
      float al = __expf(m0 - nm), p = __expf(sc0 - nm);
      l0 = l0 * al + p;
      a0.x = a0.x * al + p * vv.x; a0.y = a0.y * al + p * vv.y;
      a0.z = a0.z * al + p * vv.z; a0.w = a0.w * al + p * vv.w;
      m0 = nm;
    }
    {
      float nm = fmaxf(m1, sc1);
      float al = __expf(m1 - nm), p = __expf(sc1 - nm);
      l1 = l1 * al + p;
      a1.x = a1.x * al + p * vv.x; a1.y = a1.y * al + p * vv.y;
      a1.z = a1.z * al + p * vv.z; a1.w = a1.w * al + p * vv.w;
      m1 = nm;
    }
  }

  __shared__ float sm_m[8][2], sm_l[8][2];
  __shared__ float sm_acc[8][2][HD];
  if (lane == 0) { sm_m[wave][0] = m0; sm_l[wave][0] = l0;
                   sm_m[wave][1] = m1; sm_l[wave][1] = l1; }
  *(float4*)&sm_acc[wave][0][lane * 4] = a0;
  *(float4*)&sm_acc[wave][1][lane * 4] = a1;
  __syncthreads();

  if (wave == 0) {
#pragma unroll
    for (int g = 0; g < 2; ++g) {
      float M = sm_m[0][g];
      for (int w2 = 1; w2 < 8; ++w2) M = fmaxf(M, sm_m[w2][g]);
      float L = 0.f;
      float4 A = {0, 0, 0, 0};
      for (int w2 = 0; w2 < 8; ++w2) {
        float e = __expf(sm_m[w2][g] - M);
        L += sm_l[w2][g] * e;
        float4 aa = *(float4*)&sm_acc[w2][g][lane * 4];
        A.x += aa.x * e; A.y += aa.y * e; A.z += aa.z * e; A.w += aa.w * e;
      }
      float* pb = part + (size_t)((h * 2 + g) * nchunk + chunk) * 132;
      *(float4*)(pb + lane * 4) = A;
      if (lane == 0) { pb[HD] = M; pb[HD + 1] = L; }
    }
  }
}

// ---------------------------------------------------------------------------
// Combine partials + the s=S new-token term. Grid 16 (q-head), block 128.
// ---------------------------------------------------------------------------
__global__ void attn_combine_kernel(const float* __restrict__ part,
                                    const float* __restrict__ qr,
                                    const float* __restrict__ k_new,
                                    const float* __restrict__ v_new,
                                    const float* __restrict__ mask,
                                    float* __restrict__ attn,
                                    _Float16* __restrict__ attn_h,
                                    int S, int nchunk) {
  const int idx = blockIdx.x;   // h*2+g
  const int h   = idx >> 1;
  const int d   = threadIdx.x;  // 0..127
  __shared__ float red[4];
  __shared__ float sM, sL, sPNew;

  float p = qr[idx * HD + d] * k_new[h * HD + d];
  p = waveRedSum(p);
  if ((d & 31) == 0) red[d >> 5] = p;
  __syncthreads();
  if (d == 0) {
    float scoreNew = (red[0] + red[1] + red[2] + red[3]) * SCALE + mask[S];
    float M = scoreNew;
    for (int c = 0; c < nchunk; ++c)
      M = fmaxf(M, part[(size_t)(idx * nchunk + c) * 132 + HD]);
    float L = __expf(scoreNew - M);
    for (int c = 0; c < nchunk; ++c) {
      const float* pb = part + (size_t)(idx * nchunk + c) * 132;
      L += pb[HD + 1] * __expf(pb[HD] - M);
    }
    sM = M; sL = L; sPNew = __expf(scoreNew - M);
  }
  __syncthreads();

  float acc = sPNew * v_new[h * HD + d];
  for (int c = 0; c < nchunk; ++c) {
    const float* pb = part + (size_t)(idx * nchunk + c) * 132;
    acc += pb[d] * __expf(pb[HD] - sM);
  }
  float o = acc / sL;
  attn[idx * HD + d] = o;
  attn_h[idx * HD + d] = (_Float16)o;
}

// ---------------------------------------------------------------------------
// SiLU(gate) * up -> f16 for the down-proj WMMA GEMV.
// ---------------------------------------------------------------------------
__global__ void silu_mul_kernel(const float* __restrict__ g,
                                const float* __restrict__ u,
                                _Float16* __restrict__ outh, int n) {
  int i = blockIdx.x * blockDim.x + threadIdx.x;
  if (i < n) {
    float x = g[i];
    float s = x / (1.f + __expf(-x));
    outh[i] = (_Float16)(s * u[i]);
  }
}

// ---------------------------------------------------------------------------
extern "C" void kernel_launch(void* const* d_in, const int* in_sizes, int n_in,
                              void* d_out, int out_size, void* d_ws, size_t ws_size,
                              hipStream_t stream) {
  const float* hidden   = (const float*)d_in[0];
  const float* k_cache  = (const float*)d_in[1];
  const float* v_cache  = (const float*)d_in[2];
  const float* cosp     = (const float*)d_in[3];
  const float* sinp     = (const float*)d_in[4];
  const float* mask     = (const float*)d_in[5];
  const float* ln1_w    = (const float*)d_in[6];
  const float* q_w      = (const float*)d_in[7];
  const float* k_w      = (const float*)d_in[8];
  const float* v_w      = (const float*)d_in[9];
  const float* q_norm_w = (const float*)d_in[10];
  const float* k_norm_w = (const float*)d_in[11];
  const float* o_w      = (const float*)d_in[12];
  const float* ln2_w    = (const float*)d_in[13];
  const float* gate_w   = (const float*)d_in[14];
  const float* up_w     = (const float*)d_in[15];
  const float* down_w   = (const float*)d_in[16];

  const int S = in_sizes[1] / (NKV * HD);          // 32768
  const int nchunk = (S + CHUNK - 1) / CHUNK;      // 32

  float* out_hs = (float*)d_out;                   // [1024]
  float* k_out  = out_hs + DMODEL;                 // [NKV,(S+1),HD]
  float* v_out  = k_out + (size_t)NKV * (size_t)(S + 1) * HD;

  // Scratch layout (float units); ~360 KB total.
  float*     wsf    = (float*)d_ws;
  float*     hsn    = wsf;                               // 1024
  _Float16*  hsn_h  = (_Float16*)(wsf + 1024);           // 1024 h (512 slots)
  float*     qp     = wsf + 1536;                        // 2048
  float*     kp     = wsf + 3584;                        // 1024
  float*     vp     = wsf + 4608;                        // 1024
  float*     qr     = wsf + 5632;                        // 2048
  float*     k_new  = wsf + 7680;                        // 1024
  float*     attn   = wsf + 8704;                        // 2048
  _Float16*  attn_h = (_Float16*)(wsf + 10752);          // 2048 h (1024 slots)
  float*     hs2    = wsf + 11776;                       // 1024
  float*     hsn2   = wsf + 12800;                       // 1024
  _Float16*  hsn2_h = (_Float16*)(wsf + 13824);          // 1024 h (512 slots)
  float*     gate   = wsf + 14336;                       // 3072
  float*     up     = wsf + 17408;                       // 3072
  _Float16*  ff_h   = (_Float16*)(wsf + 20480);          // 3072 h (1536 slots)
  float*     part   = wsf + 22016;                       // 16*nchunk*132

  // 1) RMSNorm1 (+f16 copy for WMMA B operand)
  rmsnorm_kernel<<<1, 256, 0, stream>>>(hidden, ln1_w, hsn, hsn_h, DMODEL);

  // 2) QKV projections via WMMA GEMV
  gemv_wmma_kernel<<<(NH * HD) / 128, 256, 0, stream>>>(q_w, hsn_h, nullptr, qp, NH * HD, DMODEL);
  gemv_wmma_kernel<<<(NKV * HD) / 128, 256, 0, stream>>>(k_w, hsn_h, nullptr, kp, NKV * HD, DMODEL);
  gemv_wmma_kernel<<<(NKV * HD) / 128, 256, 0, stream>>>(v_w, hsn_h, nullptr, vp, NKV * HD, DMODEL);

  // 3) Per-head q/k RMSNorm + RoPE; write new k/v into cache tails
  qk_rope_kernel<<<NH + 2 * NKV, HD, 0, stream>>>(qp, kp, vp, q_norm_w, k_norm_w,
                                                  cosp, sinp, qr, k_new, k_out, v_out, S);

  // 4) Bulk cache copies (the dominant traffic besides the attention read)
  {
    long tot4 = (long)NKV * (long)S * (HD / 4);
    int blocks = (int)((tot4 + 255) / 256);
    cache_copy_kernel<<<blocks, 256, 0, stream>>>((const float4*)k_cache, (float4*)k_out, S);
    cache_copy_kernel<<<blocks, 256, 0, stream>>>((const float4*)v_cache, (float4*)v_out, S);
  }

  // 5) Flash-decoding attention over the cached S positions
  attn_partial_kernel<<<dim3(NKV, nchunk), 256, 0, stream>>>(k_cache, v_cache, qr, mask,
                                                             part, S, nchunk);
  attn_combine_kernel<<<NH, HD, 0, stream>>>(part, qr, k_new, vp, mask, attn, attn_h, S, nchunk);

  // 6) O projection + residual
  gemv_wmma_kernel<<<DMODEL / 128, 256, 0, stream>>>(o_w, attn_h, hidden, hs2, DMODEL, NH * HD);

  // 7) RMSNorm2 + MLP (gate/up WMMA GEMVs, SiLU*mul, down WMMA GEMV + residual)
  rmsnorm_kernel<<<1, 256, 0, stream>>>(hs2, ln2_w, hsn2, hsn2_h, DMODEL);
  gemv_wmma_kernel<<<FFDIM / 128, 256, 0, stream>>>(gate_w, hsn2_h, nullptr, gate, FFDIM, DMODEL);
  gemv_wmma_kernel<<<FFDIM / 128, 256, 0, stream>>>(up_w, hsn2_h, nullptr, up, FFDIM, DMODEL);
  silu_mul_kernel<<<FFDIM / 256, 256, 0, stream>>>(gate, up, ff_h, FFDIM);
  gemv_wmma_kernel<<<DMODEL / 128, 256, 0, stream>>>(down_w, ff_h, hs2, out_hs, DMODEL, FFDIM);
}
